// MultiHeadedAttention_38208029066024
// MI455X (gfx1250) — matrix operand order repaired and checked by reading, not compiled, not run on hardware
//
#include <hip/hip_runtime.h>
#include <hip/hip_bf16.h>
#include <math.h>

// Problem constants (from reference)
#define Bc  8
#define Lc  512
#define Dc  512
#define Hc  8
#define DHc 64
#define ALPHAc 1.0f
#define BETAc  1.0f

typedef __attribute__((ext_vector_type(16))) __bf16 v16bf;
typedef __attribute__((ext_vector_type(8)))  __bf16 v8bf;
typedef __attribute__((ext_vector_type(4)))  __bf16 v4bf;
typedef __attribute__((ext_vector_type(8)))  float  v8f;

// ---------------------------------------------------------------------------
// Vectorized WMMA fragment loads (wave32, V_WMMA_F32_16X16X32_BF16, ISA 7.12.2)
// A lane's fragment elements are contiguous K-runs -> 128-bit DS loads.
// ---------------------------------------------------------------------------
__device__ __forceinline__ v16bf cat8(v8bf lo, v8bf hi) {
    return __builtin_shufflevector(lo, hi, 0, 1, 2, 3, 4, 5, 6, 7,
                                           8, 9, 10, 11, 12, 13, 14, 15);
}
// A tile, row-major [rows][lda] bf16, base 16B aligned, lda % 8 == 0.
__device__ __forceinline__ v16bf frag_a_rm(const __bf16* a, int lda) {
    const unsigned l = threadIdx.x & 31u;
    const __bf16* p = a + (l & 15) * lda + 8 * ((l >> 4) & 1);
    return cat8(*(const v8bf*)(p), *(const v8bf*)(p + 16));
}
// B tile staged N-major: bt[n][k] (K contiguous per column), ldb % 8 == 0.
__device__ __forceinline__ v16bf frag_b_nm(const __bf16* bt, int ldb) {
    const unsigned l = threadIdx.x & 31u;
    const __bf16* p = bt + (l & 15) * ldb + 8 * ((l >> 4) & 1);
    return cat8(*(const v8bf*)(p), *(const v8bf*)(p + 16));
}
#define WMMA_BF16(a, b, c) \
    __builtin_amdgcn_wmma_f32_16x16x32_bf16(false, (a), false, (b), (short)0, (c), false, false)

__device__ __forceinline__ v4bf cvt4(float4 f) {
    v4bf r;
    r[0] = (__bf16)f.x; r[1] = (__bf16)f.y; r[2] = (__bf16)f.z; r[3] = (__bf16)f.w;
    return r;
}
__device__ __forceinline__ v8bf cvt8(float4 a, float4 b) {
    v8bf r;
    r[0] = (__bf16)a.x; r[1] = (__bf16)a.y; r[2] = (__bf16)a.z; r[3] = (__bf16)a.w;
    r[4] = (__bf16)b.x; r[5] = (__bf16)b.y; r[6] = (__bf16)b.z; r[7] = (__bf16)b.w;
    return r;
}

// ---------------------------------------------------------------------------
// Kernel 1: Y = (X @ W^T + bias) * scale.   Block tile 32M x 64N, block 128.
// Wave w: M-half (w&1)*16, N-half (w>>1)*32 -> 2 accumulators sharing one A frag.
// HEADSPLIT: bf16 out in [B,H,L,DH]; else f32 row-major [M,N].
// ---------------------------------------------------------------------------
template <bool HEADSPLIT>
__global__ __launch_bounds__(128) void proj_kernel(
    const float* __restrict__ X, const float* __restrict__ W,
    const float* __restrict__ bias, float scale, void* out_, int M, int N, int K) {
    __shared__ __align__(16) __bf16 Xs[32][40];
    __shared__ __align__(16) __bf16 Ws[64][40];   // N-major: Ws[n][k]
    const int m0 = blockIdx.x * 32;
    const int n0 = blockIdx.y * 64;
    const int wave = threadIdx.x >> 5;
    const int mh = (wave & 1) * 16;
    const int nh = (wave >> 1) * 32;
    v8f acc0 = {}, acc1 = {};
    for (int k0 = 0; k0 < K; k0 += 32) {
        __syncthreads();
#pragma unroll
        for (int it = 0; it < 2; ++it) {          // 1024 elems of X
            const int i = threadIdx.x * 4 + it * 512;
            const int r = i >> 5, c = i & 31;
            *(v4bf*)&Xs[r][c] = cvt4(*(const float4*)&X[(size_t)(m0 + r) * K + k0 + c]);
        }
#pragma unroll
        for (int it = 0; it < 4; ++it) {          // 2048 elems of W
            const int i = threadIdx.x * 4 + it * 512;
            const int r = i >> 5, c = i & 31;
            *(v4bf*)&Ws[r][c] = cvt4(*(const float4*)&W[(size_t)(n0 + r) * K + k0 + c]);
        }
        __syncthreads();
        const v16bf a  = frag_a_rm(&Xs[mh][0], 40);
        const v16bf b0 = frag_b_nm(&Ws[nh][0], 40);
        const v16bf b1 = frag_b_nm(&Ws[nh + 16][0], 40);
        acc0 = WMMA_BF16(a, b0, acc0);
        acc1 = WMMA_BF16(a, b1, acc1);
    }
    const unsigned l = threadIdx.x & 31u;
    const int col = l & 15, rb = 8 * (l >> 4);
#pragma unroll
    for (int i = 0; i < 8; ++i) {
        const int m = m0 + mh + i + rb;
#pragma unroll
        for (int t = 0; t < 2; ++t) {
            const int n = n0 + nh + t * 16 + col;
            const float v = (((t == 0) ? acc0[i] : acc1[i]) + bias[n]) * scale;
            if (HEADSPLIT) {
                const int bb = m >> 9, lp = m & (Lc - 1), h = n >> 6, dh = n & (DHc - 1);
                ((__bf16*)out_)[(((size_t)bb * Hc + h) * Lc + lp) * DHc + dh] = (__bf16)v;
            } else {
                ((float*)out_)[(size_t)m * N + n] = v;
            }
        }
    }
}

// ---------------------------------------------------------------------------
// Kernel 2: q2 = qh @ Wsk  (rows = flat (b,h,l), K=N=64) + qbsk = qh·bsk
// grid (B*H*L/16), block 128
// ---------------------------------------------------------------------------
__global__ __launch_bounds__(128) void q2_kernel(
    const __bf16* __restrict__ qh, const float* __restrict__ Wsk,
    const float* __restrict__ bsk, __bf16* __restrict__ q2,
    float* __restrict__ qbsk) {
    __shared__ __align__(16) __bf16 As[16][72];
    __shared__ __align__(16) __bf16 Bs[64][72];   // N-major: Bs[n][e] = Wsk[e][n]
    const int m0 = blockIdx.x * 16;
    const int wave = threadIdx.x >> 5;
    {
        const int i = threadIdx.x * 8;            // 1024 elems: one v8bf each
        const int r = i >> 6, c = i & 63;
        *(v8bf*)&As[r][c] = *(const v8bf*)&qh[(size_t)(m0 + r) * DHc + c];
    }
#pragma unroll
    for (int it = 0; it < 8; ++it) {              // transpose-scatter Wsk (tiny)
        const int j = threadIdx.x + it * 128;     // 1024 tasks of 4
        const int n = j & 63, e0 = (j >> 6) * 4;
        v4bf v;
#pragma unroll
        for (int e = 0; e < 4; ++e) v[e] = (__bf16)Wsk[(e0 + e) * DHc + n];
        *(v4bf*)&Bs[n][e0] = v;
    }
    __syncthreads();
    v8f acc = {};
#pragma unroll
    for (int kk = 0; kk < 64; kk += 32)
        acc = WMMA_BF16(frag_a_rm(&As[0][kk], 72), frag_b_nm(&Bs[wave * 16][kk], 72), acc);
    const unsigned l = threadIdx.x & 31u;
    const int col = l & 15, rb = 8 * (l >> 4);
#pragma unroll
    for (int i = 0; i < 8; ++i)
        q2[(size_t)(m0 + i + rb) * DHc + wave * 16 + col] = (__bf16)acc[i];
    if (threadIdx.x < 16) {
        float s = 0.f;
        for (int e = 0; e < DHc; ++e) s += (float)As[threadIdx.x][e] * bsk[e];
        qbsk[m0 + threadIdx.x] = s;
    }
}

// ---------------------------------------------------------------------------
// Kernel 3: scores[bh,q,k] = qh[bh,q,:]·kh[bh,k,:] + qbsk[bh,q]
// Block tile 32q x 64k, K=64 fully staged. grid (L/32, L/64, B*H), block 128.
// ---------------------------------------------------------------------------
__global__ __launch_bounds__(128) void qk_scores_kernel(
    const __bf16* __restrict__ qh, const __bf16* __restrict__ kh,
    const float* __restrict__ qbsk, float* __restrict__ scores) {
    __shared__ __align__(16) __bf16 As[32][72];
    __shared__ __align__(16) __bf16 Bs[64][72];   // N-major: Bs[kpos][d] = kh natural
    const int bh = blockIdx.z;
    const int q0 = blockIdx.x * 32;
    const int n0 = blockIdx.y * 64;
    const __bf16* qrow = qh + ((size_t)bh * Lc + q0) * DHc;
    const __bf16* krow = kh + ((size_t)bh * Lc + n0) * DHc;
#pragma unroll
    for (int it = 0; it < 2; ++it) {              // 2048 elems of A
        const int i = threadIdx.x * 8 + it * 1024;
        const int r = i >> 6, c = i & 63;
        *(v8bf*)&As[r][c] = *(const v8bf*)&qrow[(size_t)r * DHc + c];
    }
#pragma unroll
    for (int it = 0; it < 4; ++it) {              // 4096 elems of B
        const int j = threadIdx.x + it * 128;     // 512 v8 tasks
        const int n = j >> 3, c = (j & 7) * 8;
        *(v8bf*)&Bs[n][c] = *(const v8bf*)&krow[(size_t)n * DHc + c];
    }
    __syncthreads();
    const int wave = threadIdx.x >> 5;
    const int mh = (wave & 1) * 16;
    const int nh = (wave >> 1) * 32;
    v8f acc0 = {}, acc1 = {};
#pragma unroll
    for (int kk = 0; kk < 64; kk += 32) {
        const v16bf a  = frag_a_rm(&As[mh][kk], 72);
        const v16bf b0 = frag_b_nm(&Bs[nh][kk], 72);
        const v16bf b1 = frag_b_nm(&Bs[nh + 16][kk], 72);
        acc0 = WMMA_BF16(a, b0, acc0);
        acc1 = WMMA_BF16(a, b1, acc1);
    }
    const unsigned l = threadIdx.x & 31u;
    const int col = l & 15, rb = 8 * (l >> 4);
#pragma unroll
    for (int i = 0; i < 8; ++i) {
        const int m = q0 + mh + i + rb;
        const float qb = qbsk[(size_t)bh * Lc + m];
        scores[((size_t)bh * Lc + m) * Lc + n0 + nh + col]      = acc0[i] + qb;
        scores[((size_t)bh * Lc + m) * Lc + n0 + nh + 16 + col] = acc1[i] + qb;
    }
}

// ---------------------------------------------------------------------------
// Kernel 4: scores[b,h,q,k] += ALPHA * q2[b,h,q,:]·structure[b,q,k,:]
// (sk never materialized). 4 waves x 2 N-subtiles over 128 kpos, shared A frag.
// grid (L/128, B*L), block 128
// ---------------------------------------------------------------------------
__global__ __launch_bounds__(128) void struct_scores_kernel(
    const float* __restrict__ structure, const __bf16* __restrict__ q2,
    float* __restrict__ scores) {
    __shared__ __align__(16) __bf16 As[16][72];
    __shared__ __align__(16) __bf16 Bs[128][72];  // N-major: Bs[kpos][d] = S natural
    const int bq = blockIdx.y;
    const int b = bq >> 9, q = bq & (Lc - 1);
    const int n0 = blockIdx.x * 128;
    {
        const int i = threadIdx.x * 8;            // 1024 elems of A (pad h>=8 with 0)
        const int h = i >> 6, d = i & 63;
        v8bf v = {};
        if (h < Hc) v = *(const v8bf*)&q2[(((size_t)b * Hc + h) * Lc + q) * DHc + d];
        *(v8bf*)&As[h][d] = v;
    }
    const float* S = structure + (((size_t)b * Lc + q) * Lc + n0) * DHc;
    __builtin_prefetch(S + (size_t)(threadIdx.x & 31u) * DHc, 0, 1);  // global_prefetch_b8
#pragma unroll
    for (int it = 0; it < 8; ++it) {              // 8192 elems, v8 per task
        const int j = threadIdx.x + it * 128;     // 1024 tasks
        const int kl = j >> 3, d0 = (j & 7) * 8;
        const float4 f0 = *(const float4*)&S[(size_t)kl * DHc + d0];
        const float4 f1 = *(const float4*)&S[(size_t)kl * DHc + d0 + 4];
        *(v8bf*)&Bs[kl][d0] = cvt8(f0, f1);
    }
    __syncthreads();
    const int wave = threadIdx.x >> 5;
    const int nh = wave * 32;
    v8f acc0 = {}, acc1 = {};
#pragma unroll
    for (int kk = 0; kk < 64; kk += 32) {
        const v16bf a  = frag_a_rm(&As[0][kk], 72);
        const v16bf b0 = frag_b_nm(&Bs[nh][kk], 72);
        const v16bf b1 = frag_b_nm(&Bs[nh + 16][kk], 72);
        acc0 = WMMA_BF16(a, b0, acc0);
        acc1 = WMMA_BF16(a, b1, acc1);
    }
    const unsigned l = threadIdx.x & 31u;
    const int col = l & 15, g = l >> 4;
#pragma unroll
    for (int i = 0; i < 8; ++i) {
        const int h = i + 8 * g;
        if (h < Hc) {
            float* sr = scores + (((size_t)b * Hc + h) * Lc + q) * Lc + n0 + nh + col;
            sr[0]  += ALPHAc * acc0[i];
            sr[16] += ALPHAc * acc1[i];
        }
    }
}

// ---------------------------------------------------------------------------
// Kernel 5: row softmax (in place) + emit attn[:,0] to d_out tail.
// grid (B*H*L), block 128
// ---------------------------------------------------------------------------
__global__ __launch_bounds__(128) void softmax_kernel(
    float* __restrict__ scores, float* __restrict__ first_head) {
    __shared__ float red[128];
    const size_t row = blockIdx.x;                // (b*H + h)*L + q
    float* r = scores + row * Lc;
    float m = -INFINITY;
    for (int i = threadIdx.x; i < Lc; i += 128) m = fmaxf(m, r[i]);
    red[threadIdx.x] = m;
    __syncthreads();
    for (int s = 64; s > 0; s >>= 1) {
        if ((int)threadIdx.x < s) red[threadIdx.x] = fmaxf(red[threadIdx.x], red[threadIdx.x + s]);
        __syncthreads();
    }
    m = red[0];
    __syncthreads();
    float sum = 0.f;
    for (int i = threadIdx.x; i < Lc; i += 128) {
        const float e = __expf(r[i] - m);
        r[i] = e;
        sum += e;
    }
    red[threadIdx.x] = sum;
    __syncthreads();
    for (int s = 64; s > 0; s >>= 1) {
        if ((int)threadIdx.x < s) red[threadIdx.x] += red[threadIdx.x + s];
        __syncthreads();
    }
    const float inv = 1.0f / red[0];
    const int h = (int)((row >> 9) & (Hc - 1));
    const size_t b = row / ((size_t)Hc * Lc);
    const int q = (int)(row & (Lc - 1));
    for (int i = threadIdx.x; i < Lc; i += 128) {
        const float a = r[i] * inv;
        r[i] = a;
        if (h == 0) first_head[((size_t)b * Lc + q) * Lc + i] = a;
    }
}

// ---------------------------------------------------------------------------
// Kernel 6: ctx[b,q,h*64+d] = attn[b,h,q,:] @ vh[b,h,:,d]
// Block tile 32q x 64d; wave = (M-half, N-half(32)) with shared A frag.
// grid (L/32, B*H), block 128
// ---------------------------------------------------------------------------
__global__ __launch_bounds__(128) void ctx_v_kernel(
    const float* __restrict__ attn, const __bf16* __restrict__ vh,
    float* __restrict__ ctx) {
    __shared__ __align__(16) __bf16 As[32][40];
    __shared__ __align__(16) __bf16 Bt[64][40];   // N-major: Bt[d][kpos] (transposed vh)
    const int bh = blockIdx.y;
    const int b = bh >> 3, h = bh & (Hc - 1);
    const int q0 = blockIdx.x * 32;
    const float* arow = attn + ((size_t)bh * Lc + q0) * Lc;
    const __bf16* vrow = vh + (size_t)bh * Lc * DHc;
    const int wave = threadIdx.x >> 5;
    const int mh = (wave & 1) * 16;
    const int nh = (wave >> 1) * 32;
    v8f acc0 = {}, acc1 = {};
    for (int k0 = 0; k0 < Lc; k0 += 32) {
        __syncthreads();
#pragma unroll
        for (int it = 0; it < 2; ++it) {          // 1024 f32 attn elems
            const int i = threadIdx.x * 4 + it * 512;
            const int r = i >> 5, c = i & 31;
            *(v4bf*)&As[r][c] = cvt4(*(const float4*)&arow[(size_t)r * Lc + k0 + c]);
        }
#pragma unroll
        for (int it = 0; it < 2; ++it) {          // 256 tasks: v8 load + 8 scatter
            const int j = threadIdx.x + it * 128;
            const int kl = j >> 3, d0 = (j & 7) * 8;
            const v8bf v = *(const v8bf*)&vrow[(size_t)(k0 + kl) * DHc + d0];
#pragma unroll
            for (int e = 0; e < 8; ++e) Bt[d0 + e][kl] = v[e];
        }
        __syncthreads();
        const v16bf a  = frag_a_rm(&As[mh][0], 40);
        const v16bf b0 = frag_b_nm(&Bt[nh][0], 40);
        const v16bf b1 = frag_b_nm(&Bt[nh + 16][0], 40);
        acc0 = WMMA_BF16(a, b0, acc0);
        acc1 = WMMA_BF16(a, b1, acc1);
    }
    const unsigned l = threadIdx.x & 31u;
    const int col = l & 15, rb = 8 * (l >> 4);
#pragma unroll
    for (int i = 0; i < 8; ++i) {
        const int q = q0 + mh + i + rb;
        float* cr = ctx + ((size_t)b * Lc + q) * Dc + h * DHc + nh + col;
        cr[0]  = acc0[i];
        cr[16] = acc1[i];
    }
}

// ---------------------------------------------------------------------------
// Kernel 7: t[h,:] = attn[b,h,q,:] @ structure[b,q,:,:], then
// ctx += BETA*(t @ Wsv^T + bsv)  (sv never materialized; sum(attn)=1).
// grid (B*L), block 128
// ---------------------------------------------------------------------------
__global__ __launch_bounds__(128) void ctx_struct_kernel(
    const float* __restrict__ attn, const float* __restrict__ structure,
    const float* __restrict__ Wsv, const float* __restrict__ bsv,
    float* __restrict__ ctx) {
    __shared__ __align__(16) __bf16 As[16][40];
    __shared__ __align__(16) __bf16 Bt[64][40];   // N-major: Bt[d][kpos]
    __shared__ float Ts[16][DHc];
    const int bq = blockIdx.x;
    const int b = bq >> 9, q = bq & (Lc - 1);
    const float* S = structure + ((size_t)b * Lc + q) * Lc * DHc;
    const int wave = threadIdx.x >> 5;
    v8f acc = {};
    for (int k0 = 0; k0 < Lc; k0 += 32) {
        __syncthreads();
        __builtin_prefetch(S + (size_t)(k0 + 32 + (threadIdx.x & 31u)) * DHc, 0, 1);
        {
            const int i = threadIdx.x * 4;
            const int h = i >> 5, c = i & 31;
            v4bf av = {};
            if (h < Hc)
                av = cvt4(*(const float4*)&attn[(((size_t)b * Hc + h) * Lc + q) * Lc + k0 + c]);
            *(v4bf*)&As[h][c] = av;
        }
#pragma unroll
        for (int it = 0; it < 2; ++it) {          // 256 tasks: 2x float4 + 8 scatter
            const int j = threadIdx.x + it * 128;
            const int kl = j >> 3, d0 = (j & 7) * 8;
            const float4 f0 = *(const float4*)&S[(size_t)(k0 + kl) * DHc + d0];
            const float4 f1 = *(const float4*)&S[(size_t)(k0 + kl) * DHc + d0 + 4];
            const v8bf v = cvt8(f0, f1);
#pragma unroll
            for (int e = 0; e < 8; ++e) Bt[d0 + e][kl] = v[e];
        }
        __syncthreads();
        acc = WMMA_BF16(frag_a_rm(&As[0][0], 40), frag_b_nm(&Bt[wave * 16][0], 40), acc);
    }
    const unsigned l = threadIdx.x & 31u;
    const int col = l & 15, g = l >> 4;
    __syncthreads();
#pragma unroll
    for (int i = 0; i < 8; ++i) Ts[i + 8 * g][wave * 16 + col] = acc[i];
    __syncthreads();
    for (int i = threadIdx.x; i < Hc * DHc; i += 128) {
        const int h = i >> 6, d = i & 63;
        float s = bsv[d];
        for (int e = 0; e < DHc; ++e) s += Wsv[d * DHc + e] * Ts[h][e];
        ctx[((size_t)b * Lc + q) * Dc + h * DHc + d] += BETAc * s;
    }
}

// ---------------------------------------------------------------------------
extern "C" void kernel_launch(void* const* d_in, const int* in_sizes, int n_in,
                              void* d_out, int out_size, void* d_ws, size_t ws_size,
                              hipStream_t stream) {
    (void)in_sizes; (void)n_in; (void)out_size; (void)ws_size;
    const float* query = (const float*)d_in[0];
    const float* key   = (const float*)d_in[1];
    const float* value = (const float*)d_in[2];
    const float* structure = (const float*)d_in[3];
    const float* Wq = (const float*)d_in[4];  const float* bq = (const float*)d_in[5];
    const float* Wk = (const float*)d_in[6];  const float* bk = (const float*)d_in[7];
    const float* Wv = (const float*)d_in[8];  const float* bv = (const float*)d_in[9];
    const float* Wsk = (const float*)d_in[10]; const float* bsk = (const float*)d_in[11];
    const float* Wsv = (const float*)d_in[12]; const float* bsv = (const float*)d_in[13];
    const float* Wo = (const float*)d_in[14]; const float* bo = (const float*)d_in[15];
    float* out = (float*)d_out;

    // workspace carve-out
    char* ws = (char*)d_ws;
    size_t p = 0;
    const size_t nQ = (size_t)Bc * Hc * Lc * DHc;            // 2,097,152
    __bf16* qh = (__bf16*)(ws + p); p += nQ * 2;             // 4 MiB each
    __bf16* kh = (__bf16*)(ws + p); p += nQ * 2;
    __bf16* vh = (__bf16*)(ws + p); p += nQ * 2;
    __bf16* q2 = (__bf16*)(ws + p); p += nQ * 2;
    float* qbsk = (float*)(ws + p); p += (size_t)Bc * Hc * Lc * 4;
    float* scores = (float*)(ws + p); p += (size_t)Bc * Hc * Lc * Lc * 4;  // 64 MiB
    float* ctx = (float*)(ws + p); p += (size_t)Bc * Lc * Dc * 4;          // 8 MiB

    const dim3 blk(128);
    const dim3 gP(Bc * Lc / 32, Dc / 64);                    // 128 x 8
    const float qscale = 1.0f / sqrtf((float)DHc);           // 0.125

    proj_kernel<true><<<gP, blk, 0, stream>>>(query, Wq, bq, qscale, qh, Bc * Lc, Dc, Dc);
    proj_kernel<true><<<gP, blk, 0, stream>>>(key,   Wk, bk, 1.0f,  kh, Bc * Lc, Dc, Dc);
    proj_kernel<true><<<gP, blk, 0, stream>>>(value, Wv, bv, 1.0f,  vh, Bc * Lc, Dc, Dc);
    q2_kernel<<<dim3(Bc * Hc * Lc / 16), blk, 0, stream>>>(qh, Wsk, bsk, q2, qbsk);
    qk_scores_kernel<<<dim3(Lc / 32, Lc / 64, Bc * Hc), blk, 0, stream>>>(qh, kh, qbsk, scores);
    struct_scores_kernel<<<dim3(Lc / 128, Bc * Lc), blk, 0, stream>>>(structure, q2, scores);
    softmax_kernel<<<dim3(Bc * Hc * Lc), blk, 0, stream>>>(scores, out + (size_t)Bc * Lc * Dc);
    ctx_v_kernel<<<dim3(Lc / 32, Bc * Hc), blk, 0, stream>>>(scores, vh, ctx);
    ctx_struct_kernel<<<dim3(Bc * Lc), blk, 0, stream>>>(scores, structure, Wsv, bsv, ctx);
    proj_kernel<false><<<gP, blk, 0, stream>>>(ctx, Wo, bo, 1.0f, out, Bc * Lc, Dc, Dc);
}
